// VerMindAttention_36146444763900
// MI455X (gfx1250) — matrix operand order repaired and checked by reading, not compiled, not run on hardware
//
#include <hip/hip_runtime.h>

typedef __attribute__((ext_vector_type(16))) _Float16 v16h;
typedef __attribute__((ext_vector_type(8)))  _Float16 v8h;
typedef __attribute__((ext_vector_type(8)))  float    v8f;

#define B_    2
#define S_    2048
#define HID_  2048
#define NH_   32
#define NKV_  8
#define HD_   64
#define GROUP_ 4
#define QKVW_ ((NH_ + 2 * NKV_) * HD_)   /* 3072 */
#define SCALE_ 0.125f                    /* 64^-0.5 */

// ---------------------------------------------------------------------------
// CDNA5 WMMA fragment layouts (CDNA5 ISA §7.12.2, wave32), stored as
// 32 lanes x 16 f16 (512 f16 = 1 KB) per 16x32 A-tile / 32x16 B-tile.
//   A (16x32 MxK): lane = M + 16*((K&15)>=8); within lane, f16 slot
//                  = (K&7) + 8*(K>=16)  -> 8 consecutive K are 16 contiguous B
//   B (32x16 KxN): lane = N + 16*((K&15)>=8); slot = (K&7) + 8*(K>=16) -> 8 consecutive K contiguous
//   C/D (16x16):   lane holds N = lane&15; vgpr v holds M = v + 8*(lane>>4)
// ---------------------------------------------------------------------------
__device__ __forceinline__ int a_off(int row, int kk) {
  int lane = row + (((kk & 15) >= 8) ? 16 : 0);
  return lane * 16 + (kk & 7) + ((kk >= 16) ? 8 : 0);
}
__device__ __forceinline__ int b_off(int kk, int col) {
  int lane = col + (((kk & 15) >= 8) ? 16 : 0);
  return lane * 16 + (kk & 7) + ((kk >= 16) ? 8 : 0);
}

// ---------------------------------------------------------------------------
// GEMM: C[M,N] = A[M,K] @ B[K,N], fp32 in/out, f16 WMMA compute.
// 256 threads = 8 waves. Tile BM=128, BN=64, BK=32; wave w -> rows [16w,16w+16).
// Register-staged global->LDS with ping-pong buffers, 1 barrier/step.
// ---------------------------------------------------------------------------
__global__ __launch_bounds__(256) void gemm_f16wmma(const float* __restrict__ A,
                                                    const float* __restrict__ B,
                                                    float* __restrict__ C,
                                                    int M, int N, int K) {
  __shared__ __align__(16) _Float16 Asw[2][8 * 512];   // 8 m-tiles, A-frag order
  __shared__ __align__(16) _Float16 Bsw[2][4 * 512];   // 4 n-tiles, B-frag order
  const int tid  = threadIdx.x;
  const int lane = tid & 31, wave = tid >> 5;
  const int bm = blockIdx.y * 128, bn = blockIdx.x * 64;

  // A staging: 512 chunks of (1 row x 8 K); thread handles chunks tid, tid+256.
  const int am0 = tid >> 2,          akc0 = (tid & 3) * 8;
  const int am1 = (tid + 256) >> 2,  akc1 = ((tid + 256) & 3) * 8;
  // B staging: thread handles column n = tid&63, K range [(tid>>6)*8, +8).
  const int bnn = tid & 63,          bkc  = (tid >> 6) * 8;

  float aR[2][8];
  float bR[8];

  auto load_regs = [&](int k0) {
    const float4* p0 = (const float4*)(A + (size_t)(bm + am0) * K + k0 + akc0);
    const float4* p1 = (const float4*)(A + (size_t)(bm + am1) * K + k0 + akc1);
    float4 x0 = p0[0], x1 = p0[1], y0 = p1[0], y1 = p1[1];
    aR[0][0]=x0.x; aR[0][1]=x0.y; aR[0][2]=x0.z; aR[0][3]=x0.w;
    aR[0][4]=x1.x; aR[0][5]=x1.y; aR[0][6]=x1.z; aR[0][7]=x1.w;
    aR[1][0]=y0.x; aR[1][1]=y0.y; aR[1][2]=y0.z; aR[1][3]=y0.w;
    aR[1][4]=y1.x; aR[1][5]=y1.y; aR[1][6]=y1.z; aR[1][7]=y1.w;
    const float* bp = B + (size_t)(k0 + bkc) * N + bn + bnn;
#pragma unroll
    for (int j = 0; j < 8; ++j) bR[j] = bp[(size_t)j * N];   // coalesced over lanes
  };
  auto store_lds = [&](int buf) {
    v8h h0, h1, hb;
#pragma unroll
    for (int j = 0; j < 8; ++j) { h0[j] = (_Float16)aR[0][j];
                                  h1[j] = (_Float16)aR[1][j];
                                  hb[j] = (_Float16)bR[j]; }
    *(v8h*)&Asw[buf][(am0 >> 4) * 512 + a_off(am0 & 15, akc0)] = h0;
    *(v8h*)&Asw[buf][(am1 >> 4) * 512 + a_off(am1 & 15, akc1)] = h1;
    *(v8h*)&Bsw[buf][(bnn >> 4) * 512 + b_off(bkc, bnn & 15)]  = hb;
  };

  v8f acc[4] = {};
  load_regs(0);
  store_lds(0);

  const int nsteps = K >> 5;
  for (int step = 0; step < nsteps; ++step) {
    const int cur = step & 1;
    __syncthreads();                               // cur buffer ready
    if (step + 1 < nsteps) load_regs((step + 1) << 5);   // loads fly over WMMAs
    v16h af = *(const v16h*)&Asw[cur][wave * 512 + lane * 16];
#pragma unroll
    for (int nt = 0; nt < 4; ++nt) {
      v16h bf = *(const v16h*)&Bsw[cur][nt * 512 + lane * 16];
      acc[nt] = __builtin_amdgcn_wmma_f32_16x16x32_f16(
          false, af, false, bf, (short)0, acc[nt], false, false);
    }
    if (step + 1 < nsteps) store_lds(cur ^ 1);     // other buffer: no extra barrier
  }

  const int half = lane >> 4, col = lane & 15;
#pragma unroll
  for (int nt = 0; nt < 4; ++nt)
#pragma unroll
    for (int v = 0; v < 8; ++v) {
      int m = bm + wave * 16 + v + 8 * half;
      C[(size_t)m * N + bn + nt * 16 + col] = acc[nt][v];
    }
}

// ---------------------------------------------------------------------------
// RoPE + pack Q/K/V into global fragment-order f16 buffers.
// Q: A-tiles (16 q x 32 hd), tile = ((b*NH+h)*(S/16)+(q>>4))*2 + (d>>5)
// K: B-tiles (32 hd x 16 keys), tile = ((b*NKV+kv)*(S/16)+(s>>4))*2 + (d>>5)
// V: B-tiles (32 keys x 16 hd), tile = ((b*NKV+kv)*(S/32)+(s>>5))*4 + (d>>4)
// ---------------------------------------------------------------------------
__global__ __launch_bounds__(256) void rope_pack(const float* __restrict__ qkv,
                                                 const int* __restrict__ positions,
                                                 _Float16* __restrict__ Qa,
                                                 _Float16* __restrict__ Kb,
                                                 _Float16* __restrict__ Vb) {
  size_t idx = (size_t)blockIdx.x * 256 + threadIdx.x;   // over B*S*3072
  int col = (int)(idx % QKVW_);
  size_t r = idx / QKVW_;                                // b*S + s
  int s = (int)(r % S_);
  int b = (int)(r / S_);
  const float* row = qkv + r * QKVW_;
  float pos = (float)positions[r];

  if (col < NH_ * HD_) {                                 // ---- Q + RoPE
    int h = col >> 6, d = col & 63, d2 = d & 31;
    float fr = pos * __powf(10000.0f, -(float)d2 * (1.0f / 32.0f));
    float c = __cosf(fr), sn = __sinf(fr);
    float x = row[(h << 6) + d];
    float val = (d < 32) ? (x * c - row[(h << 6) + d + 32] * sn)
                         : (x * c + row[(h << 6) + d - 32] * sn);
    size_t tile = ((size_t)(b * NH_ + h) * (S_ >> 4) + (s >> 4)) * 2 + (d >> 5);
    Qa[tile * 512 + a_off(s & 15, d & 31)] = (_Float16)val;
  } else if (col < (NH_ + NKV_) * HD_) {                 // ---- K + RoPE
    int cc = col - NH_ * HD_;
    int kv = cc >> 6, d = cc & 63, d2 = d & 31;
    const float* kr = row + NH_ * HD_ + (kv << 6);
    float fr = pos * __powf(10000.0f, -(float)d2 * (1.0f / 32.0f));
    float c = __cosf(fr), sn = __sinf(fr);
    float x = kr[d];
    float val = (d < 32) ? (x * c - kr[d + 32] * sn)
                         : (x * c + kr[d - 32] * sn);
    size_t tile = ((size_t)(b * NKV_ + kv) * (S_ >> 4) + (s >> 4)) * 2 + (d >> 5);
    Kb[tile * 512 + b_off(d & 31, s & 15)] = (_Float16)val;
  } else {                                               // ---- V passthrough
    int cc = col - (NH_ + NKV_) * HD_;
    int kv = cc >> 6, d = cc & 63;
    float x = row[(NH_ + NKV_) * HD_ + (kv << 6) + d];
    size_t tile = ((size_t)(b * NKV_ + kv) * (S_ >> 5) + (s >> 5)) * 4 + (d >> 4);
    Vb[tile * 512 + b_off(s & 31, d & 15)] = (_Float16)x;
  }
}

// ---------------------------------------------------------------------------
// Causal GQA flash attention. Grid: (S/64, B*NH), block 128 = 4 waves.
// Wave w owns 16 query rows. Key step = 32: 4 WMMAs QK^T, online softmax
// (shfl_xor row reductions stay inside 16-lane halves), P -> per-wave LDS
// (D-layout -> A-layout), 4 WMMAs PV. V fragments are loaded immediately
// after the score WMMAs so their latency overlaps the softmax VALU work.
// No __syncthreads in the divergent key loop (per-wave-private LDS only).
// ---------------------------------------------------------------------------
__global__ __launch_bounds__(128) void flash_kernel(const _Float16* __restrict__ Qa,
                                                    const _Float16* __restrict__ Kb,
                                                    const _Float16* __restrict__ Vb,
                                                    float* __restrict__ attn) {
  __shared__ __align__(32) _Float16 Plds[4][512];
  const int lane = threadIdx.x & 31, wave = threadIdx.x >> 5;
  const int qtile = blockIdx.x;
  const int bh = blockIdx.y;
  const int b = bh / NH_, h = bh % NH_;
  const int kv = h / GROUP_;
  const int qw = qtile * 64 + wave * 16;
  const int col = lane & 15, half = lane >> 4;

  size_t qbase = ((size_t)(b * NH_ + h) * (S_ >> 4) + (qw >> 4)) * 2;
  v16h qa0 = *(const v16h*)(Qa + qbase * 512 + lane * 16);
  v16h qa1 = *(const v16h*)(Qa + (qbase + 1) * 512 + lane * 16);

  float mrow[8], lrow[8];
  v8f O[4] = {};
#pragma unroll
  for (int v = 0; v < 8; ++v) { mrow[v] = -3.0e38f; lrow[v] = 0.0f; }

  const size_t kbase = (size_t)(b * NKV_ + kv) * (S_ >> 4);
  const size_t vbase = (size_t)(b * NKV_ + kv) * (S_ >> 5);
  _Float16* pw = Plds[wave];

  const int kend = qw + 16;                      // exclusive key bound
  for (int k0 = 0; k0 < kend; k0 += 32) {
    const int t16 = k0 >> 4;
    const _Float16* kp = Kb + (kbase + t16) * 2 * 512 + lane * 16;
    v16h kb00 = *(const v16h*)(kp);
    v16h kb01 = *(const v16h*)(kp + 512);
    v16h kb10 = *(const v16h*)(kp + 1024);
    v16h kb11 = *(const v16h*)(kp + 1536);

    v8f s0 = {}, s1 = {};
    s0 = __builtin_amdgcn_wmma_f32_16x16x32_f16(false, qa0, false, kb00, (short)0, s0, false, false);
    s0 = __builtin_amdgcn_wmma_f32_16x16x32_f16(false, qa1, false, kb01, (short)0, s0, false, false);
    s1 = __builtin_amdgcn_wmma_f32_16x16x32_f16(false, qa0, false, kb10, (short)0, s1, false, false);
    s1 = __builtin_amdgcn_wmma_f32_16x16x32_f16(false, qa1, false, kb11, (short)0, s1, false, false);

    // Issue V-fragment loads now; latency hides under the softmax below.
    const _Float16* vp = Vb + (vbase + (size_t)(k0 >> 5)) * 4 * 512 + lane * 16;
    v16h vb0 = *(const v16h*)(vp);
    v16h vb1 = *(const v16h*)(vp + 512);
    v16h vb2 = *(const v16h*)(vp + 1024);
    v16h vb3 = *(const v16h*)(vp + 1536);

#pragma unroll
    for (int v = 0; v < 8; ++v) {
      const int qrow = qw + v + 8 * half;
      float a0 = s0[v] * SCALE_;
      float a1 = s1[v] * SCALE_;
      if (k0 + col > qrow)      a0 = -3.0e38f;   // causal mask
      if (k0 + 16 + col > qrow) a1 = -3.0e38f;
      float mx = fmaxf(a0, a1);
#pragma unroll
      for (int off = 8; off >= 1; off >>= 1)
        mx = fmaxf(mx, __shfl_xor(mx, off, 32));
      const float mnew = fmaxf(mrow[v], mx);
      const float esc = __expf(mrow[v] - mnew);
      const float p0 = __expf(a0 - mnew);
      const float p1 = __expf(a1 - mnew);
      float rs = p0 + p1;
#pragma unroll
      for (int off = 8; off >= 1; off >>= 1)
        rs += __shfl_xor(rs, off, 32);
      mrow[v] = mnew;
      lrow[v] = lrow[v] * esc + rs;
#pragma unroll
      for (int nt = 0; nt < 4; ++nt) O[nt][v] *= esc;
      const int row = v + 8 * half;              // D-layout row -> A-layout slot
      pw[a_off(row, col)]      = (_Float16)p0;
      pw[a_off(row, 16 + col)] = (_Float16)p1;
    }
    v16h pf = *(const v16h*)(pw + lane * 16);    // compiler inserts s_wait_dscnt

    O[0] = __builtin_amdgcn_wmma_f32_16x16x32_f16(false, pf, false, vb0, (short)0, O[0], false, false);
    O[1] = __builtin_amdgcn_wmma_f32_16x16x32_f16(false, pf, false, vb1, (short)0, O[1], false, false);
    O[2] = __builtin_amdgcn_wmma_f32_16x16x32_f16(false, pf, false, vb2, (short)0, O[2], false, false);
    O[3] = __builtin_amdgcn_wmma_f32_16x16x32_f16(false, pf, false, vb3, (short)0, O[3], false, false);
  }
#pragma unroll
  for (int v = 0; v < 8; ++v) {
    const int qrow = qw + v + 8 * half;
    const float inv = 1.0f / lrow[v];
    size_t base = ((size_t)b * S_ + qrow) * (NH_ * HD_) + h * HD_ + col;
#pragma unroll
    for (int nt = 0; nt < 4; ++nt)
      attn[base + nt * 16] = O[nt][v] * inv;
  }
}

// ---------------------------------------------------------------------------
extern "C" void kernel_launch(void* const* d_in, const int* in_sizes, int n_in,
                              void* d_out, int out_size, void* d_ws, size_t ws_size,
                              hipStream_t stream) {
  const int*   positions = (const int*)d_in[0];
  const float* hidden    = (const float*)d_in[1];
  const float* Wqkv      = (const float*)d_in[2];
  const float* Wo        = (const float*)d_in[3];
  float* out = (float*)d_out;

  char* ws = (char*)d_ws;
  const size_t QKV_BYTES = (size_t)B_ * S_ * QKVW_ * sizeof(float);   // 50.3 MB
  const size_t QA_BYTES  = (size_t)B_ * NH_ * S_ * HD_ * 2;           // 16.8 MB
  const size_t KB_BYTES  = (size_t)B_ * NKV_ * S_ * HD_ * 2;          // 4.2 MB
  float*    qkv  = (float*)ws;                 // consumed by rope_pack,
  float*    attn = (float*)ws;                 // then region reused for attn
  _Float16* Qa = (_Float16*)(ws + QKV_BYTES);
  _Float16* Kb = (_Float16*)(ws + QKV_BYTES + QA_BYTES);
  _Float16* Vb = (_Float16*)(ws + QKV_BYTES + QA_BYTES + KB_BYTES);

  const int M = B_ * S_;   // 4096

  // 1) qkv = hidden @ Wqkv   (4096 x 2048 x 3072)
  gemm_f16wmma<<<dim3(QKVW_ / 64, M / 128), 256, 0, stream>>>(
      hidden, Wqkv, qkv, M, QKVW_, HID_);

  // 2) RoPE + fragment packing
  {
    size_t total = (size_t)M * QKVW_;
    rope_pack<<<(unsigned)(total / 256), 256, 0, stream>>>(
        qkv, positions, Qa, Kb, Vb);
  }

  // 3) causal GQA flash attention -> attn (fp32, [B,S,NH*HD])
  flash_kernel<<<dim3(S_ / 64, B_ * NH_), 128, 0, stream>>>(Qa, Kb, Vb, attn);

  // 4) out = attn @ Wo   (4096 x 2048 x 2048)
  gemm_f16wmma<<<dim3(HID_ / 64, M / 128), 256, 0, stream>>>(
      attn, Wo, out, M, HID_, HID_);
}
